// MaskedLinearAttention_23613730193859
// MI455X (gfx1250) — matrix operand order repaired and checked by reading, not compiled, not run on hardware
//
#include <hip/hip_runtime.h>
#include <hip/hip_bf16.h>
#include <math.h>

// ---------------------------------------------------------------------------
// CDNA5 (gfx1250) WMMA helpers: v_wmma_f32_16x16x32_bf16, wave32.
// ---------------------------------------------------------------------------
typedef __attribute__((ext_vector_type(16))) __bf16 v16bf;
typedef __attribute__((ext_vector_type(8)))  __bf16 v8bf;
typedef __attribute__((ext_vector_type(4)))  __bf16 v4bf;
typedef __attribute__((ext_vector_type(2)))  __bf16 v2bf;
typedef __attribute__((ext_vector_type(8)))  float  v8f;
typedef __attribute__((ext_vector_type(4)))  float  v4f;

#define WMMA_BF16(a, b, c) \
  __builtin_amdgcn_wmma_f32_16x16x32_bf16(false, (a), false, (b), (short)0, (c), false, false)

__device__ __forceinline__ v8f zero8() {
  v8f z = {0.f, 0.f, 0.f, 0.f, 0.f, 0.f, 0.f, 0.f};
  return z;
}

__device__ __forceinline__ v16bf cat8(v8bf lo, v8bf hi) {
  return __builtin_shufflevector(lo, hi, 0, 1, 2, 3, 4, 5, 6, 7, 8, 9, 10, 11, 12, 13, 14, 15);
}

// A fragment 16x32 (MxK) from a ROW-MAJOR tile: (m,k) = lds[(row0+m)*ld + col0+k].
// Lane layout (ISA 05_wmma §7.12.2): m=lane&15, hi=lane>>4, K = (i/8)*16 + hi*8 + i%8.
// -> two contiguous 8x bf16 runs => 2x ds_load_b128.
__device__ __forceinline__ v16bf frag_a(const __bf16* lds, int row0, int col0, int ld, int lane) {
  const int m = lane & 15;
  const int hi = (lane >> 4) & 1;
  const __bf16* p = lds + (row0 + m) * ld + col0 + (hi << 3);
  return cat8(*(const v8bf*)p, *(const v8bf*)(p + 16));
}

// B fragment 32x16 (KxN) from a COLUMN-MAJOR-BY-N tile: (k,n) = lds[(n0+n)*ldk + k0+k].
// Lane layout: n=lane&15, hi=lane>>4, K = (i/8)*16 + hi*8 + i%8 -> two contiguous 8x bf16 runs => 2x ds_load_b128.
__device__ __forceinline__ v16bf frag_b_cm(const __bf16* lds, int k0, int n0, int ldk, int lane) {
  const int n = lane & 15;
  const int hi = (lane >> 4) & 1;
  const __bf16* p = lds + (n0 + n) * ldk + k0 + (hi << 3);
  return cat8(*(const v8bf*)p, *(const v8bf*)(p + 16));
}

__device__ __forceinline__ v4bf pack4(v4f v) {
  v4bf w;
  w[0] = (__bf16)v[0]; w[1] = (__bf16)v[1]; w[2] = (__bf16)v[2]; w[3] = (__bf16)v[3];
  return w;
}

// Transposed staging: read two adjacent K-rows (4 consecutive n each) and write
// four k-contiguous bf16 pairs -> 4x (v_cvt_pk_bf16_f32 + ds_store_b32).
// dst layout: [n][k] with leading dim ldk; writes dst[(nn+j)*ldk + kk .. kk+1].
__device__ __forceinline__ void stage_t2x4(__bf16* dst, int ldk, int kk, int nn,
                                           const float* src0, const float* src1) {
  const v4f a = *(const v4f*)src0;  // row kk
  const v4f b = *(const v4f*)src1;  // row kk+1
#pragma unroll
  for (int j = 0; j < 4; ++j) {
    v2bf w;
    w[0] = (__bf16)a[j];
    w[1] = (__bf16)b[j];
    *(v2bf*)(dst + (nn + j) * ldk + kk) = w;
  }
}

// ---------------------------------------------------------------------------
// Problem constants
// ---------------------------------------------------------------------------
static constexpr int BB   = 2;
static constexpr int TT   = 2048;
static constexpr int DIMM = 512;
static constexpr int HH   = 8;
static constexpr int DD   = 64;        // qk dim == v dim
static constexpr int CH   = 64;        // chunk length
static constexpr int NC   = TT / CH;   // 32 chunks
static constexpr int ROWS = BB * TT;   // 4096

// ---------------------------------------------------------------------------
// Kernel 1: Q/K/V projection (x @ W), feature map elu(x)+1 on Q,K.
// grid = (ROWS/64, 512/64, 3), block = 256 (8 waves). 64x64 tile per block.
// ---------------------------------------------------------------------------
__global__ __launch_bounds__(256) void qkv_kernel(
    const float* __restrict__ x,
    const float* __restrict__ WQ, const float* __restrict__ WK, const float* __restrict__ WV,
    float* __restrict__ qws, float* __restrict__ kws, float* __restrict__ vws) {
  __shared__ alignas(16) __bf16 sA[64 * 32];   // x tile, row-major [r][k]
  __shared__ alignas(16) __bf16 sBt[64 * 32];  // W tile, col-major-by-n [n][k]

  const int tid = threadIdx.x, lane = tid & 31, wid = tid >> 5;
  const int row0 = blockIdx.x * 64;
  const int col0 = blockIdx.y * 64;
  const int which = blockIdx.z;
  const float* __restrict__ W = (which == 0) ? WQ : (which == 1) ? WK : WV;
  float* __restrict__ out = (which == 0) ? qws : (which == 1) ? kws : vws;

  const int rw = wid & 3;   // row strip 16*rw
  const int cw = wid >> 2;  // col half 32*cw

  v8f c0 = zero8(), c1 = zero8();

  for (int k0 = 0; k0 < DIMM; k0 += 32) {
    __syncthreads();
    // stage x tile: b128 global loads -> b64 LDS stores (row-major)
#pragma unroll
    for (int i = 0; i < 2; ++i) {
      const int e = (i * 256 + tid) * 4;
      const int r = e >> 5, c = e & 31;
      const v4f xv = *(const v4f*)(&x[(size_t)(row0 + r) * DIMM + k0 + c]);
      *(v4bf*)(&sA[r * 32 + c]) = pack4(xv);
    }
    // stage W tile transposed: 16 n-quads x 16 k-pairs = 256 units, 1/thread
    {
      const int nq = tid & 15, kp = tid >> 4;
      const float* r0 = &W[(size_t)(k0 + 2 * kp) * DIMM + col0 + 4 * nq];
      stage_t2x4(sBt, 32, 2 * kp, 4 * nq, r0, r0 + DIMM);
    }
    if (k0 + 32 < DIMM) {
      __builtin_prefetch(&x[(size_t)(row0 + (tid >> 2)) * DIMM + k0 + 32], 0, 3);
      __builtin_prefetch(&W[(size_t)(k0 + 32 + (tid >> 3)) * DIMM + col0], 0, 3);
    }
    __syncthreads();

    const v16bf a  = frag_a(sA, 16 * rw, 0, 32, lane);
    const v16bf b0 = frag_b_cm(sBt, 0, 32 * cw, 32, lane);
    const v16bf b1 = frag_b_cm(sBt, 0, 32 * cw + 16, 32, lane);
    c0 = WMMA_BF16(a, b0, c0);
    c1 = WMMA_BF16(a, b1, c1);
  }

  // epilogue: feature map for Q,K; store f32
  const int n = lane & 15, mb = ((lane >> 4) & 1) << 3;
#pragma unroll
  for (int r = 0; r < 8; ++r) {
    const int grow = row0 + 16 * rw + r + mb;
    float v0 = c0[r], v1 = c1[r];
    if (which < 2) {  // elu(x)+1 = x+1 (x>0) else exp(x)
      v0 = (v0 > 0.f) ? v0 + 1.f : __expf(v0);
      v1 = (v1 > 0.f) ? v1 + 1.f : __expf(v1);
    }
    out[(size_t)grow * DIMM + col0 + 32 * cw + n]      = v0;
    out[(size_t)grow * DIMM + col0 + 32 * cw + 16 + n] = v1;
  }
}

// ---------------------------------------------------------------------------
// Kernel 2: per-chunk state tile  ckv[m][d] = sum_t V[t][m] K[t][d], plus ksum[d].
// grid = B*H*NC = 512 blocks, block = 128 (4 waves).
// ---------------------------------------------------------------------------
__global__ __launch_bounds__(128) void ktv_kernel(
    const float* __restrict__ kws, const float* __restrict__ vws,
    float* __restrict__ ckv, float* __restrict__ cks) {
  __shared__ alignas(16) __bf16 sKt[64 * 64];  // [d][t]
  __shared__ alignas(16) __bf16 sVt[64 * 64];  // [m][t]

  const int blk = blockIdx.x;
  const int chunk = blk & (NC - 1);
  const int bh = blk >> 5;
  const int b = bh >> 3, h = bh & 7;
  const int t0 = chunk * CH;
  const int tid = threadIdx.x, lane = tid & 31, wid = tid >> 5;

  // stage transposed: 32 t-pairs x 16 d-quads = 512 units, 4/thread
#pragma unroll
  for (int i = 0; i < 4; ++i) {
    const int u = i * 128 + tid;
    const int dq = u & 15, tp = u >> 4;  // tp 0..31
    const int t = 2 * tp, d0 = 4 * dq;
    const size_t g = (size_t)(b * TT + t0 + t) * DIMM + h * DD + d0;
    stage_t2x4(sKt, 64, t, d0, &kws[g], &kws[g + DIMM]);
    stage_t2x4(sVt, 64, t, d0, &vws[g], &vws[g + DIMM]);
  }
  __syncthreads();

  // Out[m][d]: A[m][t] = sVt row-major; B[t][d] = sKt col-major-by-d.
  v8f acc[4];
#pragma unroll
  for (int j = 0; j < 4; ++j) acc[j] = zero8();

  for (int kk = 0; kk < 64; kk += 32) {  // contraction over t
    const v16bf a = frag_a(sVt, 16 * wid, kk, 64, lane);
#pragma unroll
    for (int j = 0; j < 4; ++j)
      acc[j] = WMMA_BF16(a, frag_b_cm(sKt, kk, 16 * j, 64, lane), acc[j]);
  }

  float* outp = ckv + (size_t)blk * (64 * 64);
  const int n = lane & 15, mb = ((lane >> 4) & 1) << 3;
#pragma unroll
  for (int j = 0; j < 4; ++j)
#pragma unroll
    for (int r = 0; r < 8; ++r)
      outp[(16 * wid + r + mb) * 64 + 16 * j + n] = acc[j][r];

  if (tid < 64) {  // ksum[d] = sum_t K[t][d]; sKt row is k-contiguous -> b128 reads
    float s = 0.f;
#pragma unroll
    for (int tc = 0; tc < 8; ++tc) {
      const v8bf kv = *(const v8bf*)(&sKt[tid * 64 + tc * 8]);
#pragma unroll
      for (int j = 0; j < 8; ++j) s += (float)kv[j];
    }
    cks[blk * 64 + tid] = s;
  }
}

// ---------------------------------------------------------------------------
// Kernel 3: in-place exclusive prefix over chunks per (b,h). 16 blocks x 256.
// ---------------------------------------------------------------------------
__global__ __launch_bounds__(256) void scan_kernel(float* __restrict__ ckv,
                                                   float* __restrict__ cks) {
  const int bh = blockIdx.x;
  const int tid = threadIdx.x;
  float run[16];
#pragma unroll
  for (int i = 0; i < 16; ++i) run[i] = 0.f;
  float krun = 0.f;

  for (int c = 0; c < NC; ++c) {
    float* p = ckv + (size_t)(bh * NC + c) * (64 * 64);
#pragma unroll
    for (int i = 0; i < 16; ++i) {
      const int idx = i * 256 + tid;
      const float t = p[idx];
      p[idx] = run[i];
      run[i] += t;
    }
    if (tid < 64) {
      float* kp = cks + (bh * NC + c) * 64 + tid;
      const float t = *kp;
      *kp = krun;
      krun += t;
    }
  }
}

// ---------------------------------------------------------------------------
// Kernel 4: per-chunk attention + output projection.
//   S   = mask(Q Kᵀ);  z_t = 1/(rowsum(S)_t + q_t·ksum_prev)
//   Y   = (S V + Q·State) * z;   Out = Y @ Wproj -> d_out
// grid = 512 blocks, block = 128 (4 waves).
// ---------------------------------------------------------------------------
__global__ __launch_bounds__(128) void attn_kernel(
    const float* __restrict__ qws, const float* __restrict__ kws,
    const float* __restrict__ vws, const float* __restrict__ ckv,
    const float* __restrict__ cks, const float* __restrict__ Wproj,
    float* __restrict__ out) {
  __shared__ alignas(16) __bf16 sQ[64 * 64];   // [t][d]
  __shared__ alignas(16) __bf16 sK[64 * 64];   // [t][d]; later masked scores [t1][t2]
  __shared__ alignas(16) __bf16 sVt[64 * 64];  // [m][t]
  __shared__ alignas(16) __bf16 sSt[64 * 64];  // state [m][d] (ckv layout)
  __shared__ alignas(16) __bf16 sY[64 * 64];   // [t][m]
  __shared__ alignas(16) __bf16 sWt[64 * 64];  // Wproj slice [n][m]
  __shared__ float sKsum[64];
  __shared__ float sZ[64];

  const int blk = blockIdx.x;
  const int chunk = blk & (NC - 1);
  const int bh = blk >> 5;
  const int b = bh >> 3, h = bh & 7;
  const int t0 = chunk * CH;
  const int tid = threadIdx.x, lane = tid & 31, wid = tid >> 5;

  // row-major tiles: b128 loads -> b64 LDS stores
#pragma unroll
  for (int i = 0; i < 8; ++i) {
    const int e = (i * 128 + tid) * 4;
    const int t = e >> 6, d0 = e & 63;
    const size_t g = (size_t)(b * TT + t0 + t) * DIMM + h * DD + d0;
    const v4f qv = *(const v4f*)(&qws[g]);
    const v4f kv = *(const v4f*)(&kws[g]);
    const v4f sv = *(const v4f*)(&ckv[(size_t)blk * (64 * 64) + e]);
    *(v4bf*)(&sQ[e])  = pack4(qv);
    *(v4bf*)(&sK[e])  = pack4(kv);
    *(v4bf*)(&sSt[e]) = pack4(sv);  // straight copy: ckv already [m][d]
  }
  // V transposed: 32 t-pairs x 16 m-quads, 4 units/thread
#pragma unroll
  for (int i = 0; i < 4; ++i) {
    const int u = i * 128 + tid;
    const int mq = u & 15, tp = u >> 4;
    const int t = 2 * tp, m0 = 4 * mq;
    const size_t g = (size_t)(b * TT + t0 + t) * DIMM + h * DD + m0;
    stage_t2x4(sVt, 64, t, m0, &vws[g], &vws[g + DIMM]);
  }
  if (tid < 64) sKsum[tid] = cks[blk * 64 + tid];
  __syncthreads();

  // GEMM1: scores = Q @ Kᵀ. B[k=d][n=t2] = K[t2][d] => sK row-major IS col-major-by-t2.
  v8f sc[4];
#pragma unroll
  for (int j = 0; j < 4; ++j) sc[j] = zero8();
  for (int kk = 0; kk < 64; kk += 32) {
    const v16bf a = frag_a(sQ, 16 * wid, kk, 64, lane);
#pragma unroll
    for (int j = 0; j < 4; ++j)
      sc[j] = WMMA_BF16(a, frag_b_cm(sK, kk, 16 * j, 64, lane), sc[j]);
  }
  __syncthreads();  // everyone done reading sK (raw K)

  // causal mask, write bf16 scores into sK (row-major [t1][t2])
  {
    const int n = lane & 15, mb = ((lane >> 4) & 1) << 3;
#pragma unroll
    for (int j = 0; j < 4; ++j)
#pragma unroll
      for (int r = 0; r < 8; ++r) {
        const int m = 16 * wid + r + mb;  // t1
        const int nn = 16 * j + n;        // t2
        sK[m * 64 + nn] = (__bf16)((nn <= m) ? sc[j][r] : 0.f);
      }
  }
  __syncthreads();

  // denominator: z_t = 1 / (rowsum(masked scores)_t + q_t . ksum_prev)
  // masked scores are zero above the diagonal -> full-row vector sum is causal.
  if (tid < 64) {
    const int t = tid;
    float s = 0.f;
#pragma unroll
    for (int jc = 0; jc < 8; ++jc) {
      const v8bf svv = *(const v8bf*)(&sK[t * 64 + jc * 8]);
#pragma unroll
      for (int j = 0; j < 8; ++j) s += (float)svv[j];
    }
#pragma unroll
    for (int dc = 0; dc < 8; ++dc) {
      const v8bf qv = *(const v8bf*)(&sQ[t * 64 + dc * 8]);
#pragma unroll
      for (int d = 0; d < 8; ++d) s += (float)qv[d] * sKsum[dc * 8 + d];
    }
    sZ[t] = 1.f / s;
  }
  __syncthreads();

  // GEMM2+3: Y = scores @ V + Q @ State
  v8f y[4];
#pragma unroll
  for (int j = 0; j < 4; ++j) y[j] = zero8();
  for (int kk = 0; kk < 64; kk += 32) {
    const v16bf aS = frag_a(sK, 16 * wid, kk, 64, lane);
    const v16bf aQ = frag_a(sQ, 16 * wid, kk, 64, lane);
#pragma unroll
    for (int j = 0; j < 4; ++j) {
      y[j] = WMMA_BF16(aS, frag_b_cm(sVt, kk, 16 * j, 64, lane), y[j]);
      y[j] = WMMA_BF16(aQ, frag_b_cm(sSt, kk, 16 * j, 64, lane), y[j]);
    }
  }
  // scale rows by z, stash as bf16 (row-major [t][m] -> A operand of GEMM4)
  {
    const int n = lane & 15, mb = ((lane >> 4) & 1) << 3;
#pragma unroll
    for (int j = 0; j < 4; ++j)
#pragma unroll
      for (int r = 0; r < 8; ++r) {
        const int m = 16 * wid + r + mb;
        sY[m * 64 + 16 * j + n] = (__bf16)(y[j][r] * sZ[m]);
      }
  }
  __syncthreads();

  // GEMM4: Out = Y @ Wproj, 8 column tiles of 64
  for (int cc = 0; cc < 8; ++cc) {
    // stage Wproj slice transposed: 32 k-pairs x 16 n-quads, 4 units/thread
#pragma unroll
    for (int i = 0; i < 4; ++i) {
      const int u = i * 128 + tid;
      const int nq = u & 15, kp = u >> 4;
      const float* r0 = &Wproj[(size_t)(2 * kp) * DIMM + cc * 64 + 4 * nq];
      stage_t2x4(sWt, 64, 2 * kp, 4 * nq, r0, r0 + DIMM);
    }
    __syncthreads();

    v8f o[4];
#pragma unroll
    for (int j = 0; j < 4; ++j) o[j] = zero8();
    for (int kk = 0; kk < 64; kk += 32) {
      const v16bf a = frag_a(sY, 16 * wid, kk, 64, lane);
#pragma unroll
      for (int j = 0; j < 4; ++j)
        o[j] = WMMA_BF16(a, frag_b_cm(sWt, kk, 16 * j, 64, lane), o[j]);
    }

    const int n = lane & 15, mb = ((lane >> 4) & 1) << 3;
#pragma unroll
    for (int j = 0; j < 4; ++j)
#pragma unroll
      for (int r = 0; r < 8; ++r) {
        const int m = 16 * wid + r + mb;  // local t
        const size_t gi =
            ((size_t)((b * TT + t0 + m) * HH + h)) * DIMM + cc * 64 + 16 * j + n;
        out[gi] = o[j][r];
      }
    __syncthreads();
  }
}

// ---------------------------------------------------------------------------
// Host launcher
// ---------------------------------------------------------------------------
extern "C" void kernel_launch(void* const* d_in, const int* in_sizes, int n_in,
                              void* d_out, int out_size, void* d_ws, size_t ws_size,
                              hipStream_t stream) {
  (void)in_sizes; (void)n_in; (void)out_size; (void)ws_size;
  const float* x     = (const float*)d_in[0];
  const float* WQ    = (const float*)d_in[1];
  const float* WK    = (const float*)d_in[2];
  const float* WV    = (const float*)d_in[3];
  const float* Wproj = (const float*)d_in[4];
  // d_in[5] = start_pos (ignored on the training path)
  float* out = (float*)d_out;

  // workspace layout (f32): ~32.1 MB total
  float* qws = (float*)d_ws;
  float* kws = qws + (size_t)ROWS * DIMM;
  float* vws = kws + (size_t)ROWS * DIMM;
  float* ckv = vws + (size_t)ROWS * DIMM;
  float* cks = ckv + (size_t)BB * HH * NC * 64 * 64;

  qkv_kernel<<<dim3(ROWS / 64, DIMM / 64, 3), 256, 0, stream>>>(
      x, WQ, WK, WV, qws, kws, vws);
  ktv_kernel<<<BB * HH * NC, 128, 0, stream>>>(kws, vws, ckv, cks);
  scan_kernel<<<BB * HH, 256, 0, stream>>>(ckv, cks);
  attn_kernel<<<BB * HH * NC, 128, 0, stream>>>(
      qws, kws, vws, ckv, cks, Wproj, out);
}
